// StsIndividualLayer_33165737459825
// MI455X (gfx1250) — matrix operand order repaired and checked by reading, not compiled, not run on hardware
//
#include <hip/hip_runtime.h>
#include <hip/hip_bf16.h>

typedef __attribute__((ext_vector_type(16))) __bf16 v16bf;
typedef __attribute__((ext_vector_type(8)))  __bf16 v8bf;
typedef __attribute__((ext_vector_type(8)))  float  v8f;

#define B_   16
#define T_   12
#define N_   2048
#define C_   64
#define WIN_ 3
#define F_   3
#define NW_  10
#define K1_  (WIN_ * N_)   // 6144

// ---------------------------------------------------------------------------
// Prep 1: xT[b,t,c,n] = bf16(data[b,t,n,c] + temb[t,c] + semb[n,c])
// 64x64 tile transpose through LDS so the main kernel's B-fragments are
// contiguous 32B-per-lane loads.
// ---------------------------------------------------------------------------
__global__ void prep_x_kernel(const float* __restrict__ data,
                              const float* __restrict__ temb,
                              const float* __restrict__ semb,
                              __bf16* __restrict__ xT) {
  __shared__ __bf16 tile[64][65];
  const int bt = blockIdx.y;             // b*T + t
  const int t  = bt % T_;
  const int n0 = blockIdx.x * 64;
  const int tx = threadIdx.x & 63;
  const int ty = threadIdx.x >> 6;       // 0..3
  const float* dptr = data + ((size_t)bt * N_ + n0) * C_;
  #pragma unroll
  for (int r = 0; r < 64; r += 4) {
    const int n = r + ty;
    float v = dptr[(size_t)n * C_ + tx] + temb[t * C_ + tx] + semb[(n0 + n) * C_ + tx];
    tile[tx][n] = (__bf16)v;             // transposed write: tile[c][n]
  }
  __syncthreads();
  __bf16* optr = xT + (size_t)bt * C_ * N_ + n0;
  #pragma unroll
  for (int r = 0; r < 64; r += 4) {
    const int c = r + ty;
    optr[(size_t)c * N_ + tx] = tile[c][tx];
  }
}

// ---------------------------------------------------------------------------
// Prep 2: adj_mid (rows N..2N-1 of adj) -> bf16, row-major 2048 x 6144
// ---------------------------------------------------------------------------
__global__ void prep_adj_kernel(const float* __restrict__ adj,
                                __bf16* __restrict__ adjbf) {
  const int k = blockIdx.x * 256 + threadIdx.x;  // 0..6143
  const int m = blockIdx.y;                      // 0..2047
  adjbf[(size_t)m * K1_ + k] = (__bf16)adj[(size_t)(N_ + m) * K1_ + k];
}

// ---------------------------------------------------------------------------
// Prep 3: WT[w,f,d,c] = bf16(W[w,f,c,d])  (245,760 elements)
// ---------------------------------------------------------------------------
__global__ void prep_w_kernel(const float* __restrict__ W,
                              __bf16* __restrict__ WT) {
  const int i  = blockIdx.x * 256 + threadIdx.x;  // wf*8192 + d*64 + c
  const int c  = i & 63;
  const int d  = (i >> 6) & 127;
  const int wf = i >> 13;
  WT[i] = (__bf16)W[((size_t)wf * C_ + c) * (2 * C_) + d];
}

// ---------------------------------------------------------------------------
// Fused: agg = adj_mid @ x_win (bf16 WMMA, f32 acc), then per-wave epilogue
// pre[f] = agg @ W[w,f] + bias, GLU, max over f, store fp32.
// Grid: x = b*NW+w (160), y = M-tile (16 of 128 rows). Block: 256 (8 waves).
// Each wave owns a 16x64 slab of agg / out.
// ---------------------------------------------------------------------------
__global__ __launch_bounds__(256)
void fused_stgcn_kernel(const __bf16* __restrict__ adjbf,
                        const __bf16* __restrict__ xT,
                        const __bf16* __restrict__ WTb,
                        const float*  __restrict__ bias,
                        float* __restrict__ out) {
  __shared__ float aggLds[8][16][68];   // per-wave 16x64 agg slab (padded)

  const int bw  = blockIdx.x;           // 0..159
  const int b   = bw / NW_;
  const int w   = bw % NW_;
  const int wv  = threadIdx.x >> 5;     // wave 0..7
  const int l   = threadIdx.x & 31;
  const int lhi = (l >> 4) & 1;         // lane half
  const int ll  = l & 15;
  const int m0  = blockIdx.y * 128 + wv * 16;

  v8f acc[4];
  #pragma unroll
  for (int ct = 0; ct < 4; ++ct)
    #pragma unroll
    for (int i = 0; i < 8; ++i) acc[ct][i] = 0.0f;

  // A: lane ll holds row m0+ll; elements 0..7 -> K = k0 + lhi*8 + i,
  //    elements 8..15 -> K = k0 + 16 + lhi*8 + (i-8)   (ISA 16-bit A layout)
  const __bf16* aRow  = adjbf + (size_t)(m0 + ll) * K1_ + (lhi << 3);
  const __bf16* xBase = xT + (size_t)b * T_ * C_ * N_;

  for (int k0 = 0; k0 < K1_; k0 += 32) {
    const int dt = k0 >> 11;            // which time slice of the window
    const int nb = k0 & (N_ - 1);       // n offset inside slice (32-aligned)

    v8bf alo = *(const v8bf*)(aRow + k0);
    v8bf ahi = *(const v8bf*)(aRow + k0 + 16);
    v16bf af;
    #pragma unroll
    for (int i = 0; i < 8; ++i) { af[i] = alo[i]; af[i + 8] = ahi[i]; }

    __builtin_prefetch(aRow + k0 + 64, 0, 1);   // global_prefetch_b8 next A chunk

    #pragma unroll
    for (int ct = 0; ct < 4; ++ct) {
      // B: lane ll = column c; elements 0..7 -> K = nb + lhi*8 + i, 8..15 -> K = nb + 16 + lhi*8 + (i-8)
      const __bf16* bp = xBase + ((size_t)(w + dt) * C_ + ct * 16 + ll) * N_
                       + nb + (lhi << 3);
      v8bf blo = *(const v8bf*)bp;
      v8bf bhi = *(const v8bf*)(bp + 16);
      v16bf bfr;
      #pragma unroll
      for (int i = 0; i < 8; ++i) { bfr[i] = blo[i]; bfr[i + 8] = bhi[i]; }
      acc[ct] = __builtin_amdgcn_wmma_f32_16x16x32_bf16(
          false, af, false, bfr, (short)0, acc[ct], false, false);
    }
  }

  // Spill agg (C layout: M = v + lhi*8, c = ct*16 + ll) to LDS for re-layout.
  #pragma unroll
  for (int ct = 0; ct < 4; ++ct)
    #pragma unroll
    for (int v = 0; v < 8; ++v)
      aggLds[wv][v + lhi * 8][ct * 16 + ll] = acc[ct][v];
  __syncthreads();

  // Rebuild agg as WMMA A-fragments (two K=32 chunks over C_=64), bf16.
  v16bf af2[2];
  #pragma unroll
  for (int kk = 0; kk < 2; ++kk)
    #pragma unroll
    for (int i = 0; i < 16; ++i) {
      const int k = kk * 32 + ((i < 8) ? (lhi * 8 + i) : (16 + lhi * 8 + (i - 8)));
      af2[kk][i] = (__bf16)aggLds[wv][ll][k];
    }

  v8f vmax[4];
  #pragma unroll
  for (int t2 = 0; t2 < 4; ++t2)
    #pragma unroll
    for (int i = 0; i < 8; ++i) vmax[t2][i] = -3.402823466e38f;

  const __bf16* wtBase = WTb + (size_t)(w * F_) * 128 * C_;
  const float*  bBase  = bias + (size_t)(w * F_) * 128;

  for (int f = 0; f < F_; ++f) {
    v8f acc2[8];
    #pragma unroll
    for (int t2 = 0; t2 < 8; ++t2) {
      const float bv = bBase[f * 128 + t2 * 16 + ll];  // bias depends on d only
      #pragma unroll
      for (int i = 0; i < 8; ++i) acc2[t2][i] = bv;
    }
    #pragma unroll
    for (int kk = 0; kk < 2; ++kk)
      #pragma unroll
      for (int t2 = 0; t2 < 8; ++t2) {
        const __bf16* wp = wtBase + ((size_t)f * 128 + t2 * 16 + ll) * C_
                         + kk * 32 + (lhi << 3);
        v8bf wlo = *(const v8bf*)wp;
        v8bf whi = *(const v8bf*)(wp + 16);
        v16bf wfr;
        #pragma unroll
        for (int i = 0; i < 8; ++i) { wfr[i] = wlo[i]; wfr[i + 8] = whi[i]; }
        acc2[t2] = __builtin_amdgcn_wmma_f32_16x16x32_bf16(
            false, af2[kk], false, wfr, (short)0, acc2[t2], false, false);
      }
    // GLU: lhs = col-tiles 0..3 (d 0..63), rhs = tiles 4..7 (d 64..127);
    // identical lane/VGPR alignment -> pure per-register VALU, then max over f.
    #pragma unroll
    for (int t2 = 0; t2 < 4; ++t2)
      #pragma unroll
      for (int i = 0; i < 8; ++i) {
        const float lhs = acc2[t2][i];
        const float rhs = acc2[t2 + 4][i];
        const float g = lhs * (1.0f / (1.0f + __expf(-rhs)));
        vmax[t2][i] = fmaxf(vmax[t2][i], g);
      }
  }

  // out[b,w,n,c]: n = m0 + lhi*8 + v, c = t2*16 + ll (lanes 0..15 coalesce 64B)
  float* oBase = out + (size_t)bw * N_ * C_;
  #pragma unroll
  for (int t2 = 0; t2 < 4; ++t2)
    #pragma unroll
    for (int v = 0; v < 8; ++v) {
      const int n = m0 + lhi * 8 + v;
      oBase[(size_t)n * C_ + t2 * 16 + ll] = vmax[t2][v];
    }
}

// ---------------------------------------------------------------------------
extern "C" void kernel_launch(void* const* d_in, const int* in_sizes, int n_in,
                              void* d_out, int out_size, void* d_ws, size_t ws_size,
                              hipStream_t stream) {
  (void)in_sizes; (void)n_in; (void)out_size; (void)ws_size;
  const float* data = (const float*)d_in[0];
  const float* adj  = (const float*)d_in[1];
  const float* temb = (const float*)d_in[2];
  const float* semb = (const float*)d_in[3];
  const float* W    = (const float*)d_in[4];
  const float* bias = (const float*)d_in[5];
  float* out = (float*)d_out;

  char* ws = (char*)d_ws;
  __bf16* xT    = (__bf16*)ws;                          // 50,331,648 B
  __bf16* adjbf = (__bf16*)(ws + 50331648);             // 25,165,824 B
  __bf16* WTb   = (__bf16*)(ws + 75497472);             //    491,520 B

  prep_x_kernel<<<dim3(N_ / 64, B_ * T_), 256, 0, stream>>>(data, temb, semb, xT);
  prep_adj_kernel<<<dim3(K1_ / 256, N_), 256, 0, stream>>>(adj, adjbf);
  prep_w_kernel<<<dim3((NW_ * F_ * 2 * C_ * C_) / 256), 256, 0, stream>>>(W, WTb);
  fused_stgcn_kernel<<<dim3(B_ * NW_, N_ / 128), 256, 0, stream>>>(
      adjbf, xT, WTb, bias, out);
}
